// GroupedQueryAttention_59528246722830
// MI455X (gfx1250) — matrix operand order repaired and checked by reading, not compiled, not run on hardware
//
#include <hip/hip_runtime.h>
#include <hip/hip_bf16.h>

typedef __attribute__((ext_vector_type(16))) _Float16 v16h;
typedef __attribute__((ext_vector_type(8)))  _Float16 v8h;
typedef __attribute__((ext_vector_type(8)))  float    v8f;

#define WMMA_F16(a, b, c) \
    __builtin_amdgcn_wmma_f32_16x16x32_f16(false, (a), false, (b), (short)0, (c), false, false)

// ---------------------------------------------------------------------------
// Fragment loaders (CDNA5 wave32 WMMA layouts, CDNA5 ISA 7.12.2)
//  A 16x32 f16 : lane lo holds row M=lo; halves 0-7 = K (hi*8 .. hi*8+7),
//                halves 8-15 = K (16+hi*8 .. 16+hi*8+7)   -> two v8h loads
//  B 32x16 f16 : lane lo holds col N=lo; halves follow A's K order -> strided loads
//  C/D 16x16 f32: vgpr r holds M = r + 8*hi, lane lo = N
// ---------------------------------------------------------------------------
__device__ __forceinline__ v16h load_a_frag(const _Float16* __restrict__ A,
                                            int lda, int row, int k0, int hi) {
    union { v16h v; v8h h[2]; } f;
    const _Float16* p = A + (size_t)row * lda + k0 + hi * 8;
    f.h[0] = *(const v8h*)(p);
    f.h[1] = *(const v8h*)(p + 16);
    return f.v;
}

// ---------------------------------------------------------------------------
// Conversion kernels
// ---------------------------------------------------------------------------
// 4 elements per thread: float4 in, packed 4x f16 out (B128 load / B64 store)
__global__ void cvt_f32_to_f16_v4(const float* __restrict__ in,
                                  _Float16* __restrict__ out, int n4) {
    int i = blockIdx.x * blockDim.x + threadIdx.x;
    if (i < n4) {
        const float4 v = ((const float4*)in)[i];
        union { _Float16 h[4]; unsigned long long u; } o;
        o.h[0] = (_Float16)v.x;
        o.h[1] = (_Float16)v.y;
        o.h[2] = (_Float16)v.z;
        o.h[3] = (_Float16)v.w;
        ((unsigned long long*)out)[i] = o.u;
    }
}

// W is R x C (row-major, out_features x K).  Wt is C x R: Wt[k*R + r] = W[r*C + k]
// LDS-tiled 32x32 transpose: coalesced reads along k, coalesced writes along r.
__global__ __launch_bounds__(256)
void transpose_cvt_f16_tiled(const float* __restrict__ W,
                             _Float16* __restrict__ Wt, int R, int C) {
    __shared__ _Float16 tile[32][33];  // +1 pad: avoid LDS bank conflicts
    const int k0 = blockIdx.x * 32;
    const int r0 = blockIdx.y * 32;
    const int tx = threadIdx.x;        // 0..31
    const int ty = threadIdx.y;        // 0..7
#pragma unroll
    for (int yy = 0; yy < 32; yy += 8)
        tile[ty + yy][tx] = (_Float16)W[(size_t)(r0 + ty + yy) * C + k0 + tx];
    __syncthreads();
#pragma unroll
    for (int yy = 0; yy < 32; yy += 8)
        Wt[(size_t)(k0 + ty + yy) * R + r0 + tx] = tile[tx][ty + yy];
}

// ---------------------------------------------------------------------------
// Generic WMMA GEMM:  out[m,n] = sum_k A[m,k] * Bt[k,n] + bias[n]
// One wave per 64x64 tile (4x4 subtiles of 16x16), K in steps of 32.
// MODE 0: f16 row-major store
// MODE 1: rope scale, f16 row-major store               (Q projection)
// MODE 2: rope scale, store transposed Kt[b,kvh,d,t]    (K projection)
// MODE 3: f32 row-major store                           (output projection)
// ---------------------------------------------------------------------------
#define LN10000_OVER_64 0.14391156831212787f

template <int MODE>
__global__ __launch_bounds__(32)
void gemm_wmma(const _Float16* __restrict__ A, const _Float16* __restrict__ Bt,
               const float* __restrict__ bias, void* __restrict__ outp,
               int M, int N, int K) {
    const int lane = threadIdx.x;
    const int lo = lane & 15;
    const int hi = lane >> 4;
    const int n0 = blockIdx.x * 64;
    const int m0 = blockIdx.y * 64;

    v8f acc[16] = {};

    for (int k0 = 0; k0 < K; k0 += 32) {
        v16h a[4], b[4];
#pragma unroll
        for (int mt = 0; mt < 4; ++mt)
            a[mt] = load_a_frag(A, K, m0 + mt * 16 + lo, k0, hi);
        const int krow = k0 + lo + hi * 16;
#pragma unroll
        for (int nt = 0; nt < 4; ++nt)
#pragma unroll
            for (int s = 0; s < 16; ++s)
                b[nt][s] = Bt[(size_t)(k0 + hi * 8 + (s < 8 ? s : s + 8)) * N + n0 + nt * 16 + lo];
        // cover latency of next K-step's streaming B operand (global_prefetch_b8)
        __builtin_prefetch(Bt + (size_t)(krow + 32) * N + n0, 0, 1);
#pragma unroll
        for (int mt = 0; mt < 4; ++mt)
#pragma unroll
            for (int nt = 0; nt < 4; ++nt)
                acc[mt * 4 + nt] = WMMA_F16(a[mt], b[nt], acc[mt * 4 + nt]);
    }

#pragma unroll
    for (int mt = 0; mt < 4; ++mt) {
#pragma unroll
        for (int nt = 0; nt < 4; ++nt) {
            v8f c = acc[mt * 4 + nt];
            const int ncol = n0 + nt * 16 + lo;
            const float bs = bias[ncol];
#pragma unroll
            for (int r = 0; r < 8; ++r) {
                const int m = m0 + mt * 16 + r + hi * 8;
                float val = c[r] + bs;
                if (MODE == 1 || MODE == 2) {
                    // degenerate rope: val *= cos(theta) + sin(theta)
                    const int t  = m & 1023;
                    const int dd = ncol & 127;
                    const float inv = __expf(-(float)(dd & 63) * LN10000_OVER_64);
                    const float th  = (float)t * inv;
                    val *= (__cosf(th) + __sinf(th));
                }
                if (MODE == 0 || MODE == 1) {
                    ((_Float16*)outp)[(size_t)m * N + ncol] = (_Float16)val;
                } else if (MODE == 2) {
                    const int bidx = m >> 10;
                    const int t    = m & 1023;
                    const int kvh  = ncol >> 7;
                    const int dd   = ncol & 127;
                    ((_Float16*)outp)[(size_t)((bidx * 4 + kvh) * 128 + dd) * 1024 + t] =
                        (_Float16)val;
                } else { // MODE 3
                    ((float*)outp)[(size_t)m * N + ncol] = val;
                }
            }
        }
    }
}

// ---------------------------------------------------------------------------
// Flash attention (causal, GQA 16 q-heads / 4 kv-heads, head_dim = 128).
// One wave handles a 16-row query tile for one (b, h).
// Per 32-key block: 8 WMMAs for S = Q Kt, online softmax, P (16x32 f16)
// round-trips through LDS to become an A-fragment, 8 WMMAs for acc += P V.
// ---------------------------------------------------------------------------
__global__ __launch_bounds__(32)
void attn_wmma(const _Float16* __restrict__ qh, const _Float16* __restrict__ kt,
               const _Float16* __restrict__ vh, _Float16* __restrict__ yh) {
    const int lane = threadIdx.x;
    const int lo = lane & 15;
    const int hi = lane >> 4;
    const int t0 = blockIdx.x * 16;
    const int h  = blockIdx.y;
    const int b  = blockIdx.z;
    const int kvh = h >> 2;

    __shared__ _Float16 pl[16 * 40];  // 16 rows x 32 keys, stride 40 (16B aligned)

    const _Float16* qbase = qh + (size_t)(b * 1024) * 2048 + h * 128;  // rows stride 2048
    const _Float16* ktb   = kt + (size_t)(b * 4 + kvh) * 128 * 1024;   // [d][t]
    const _Float16* vb    = vh + (size_t)(b * 1024) * 512 + kvh * 128; // [t][d], stride 512

    v8f acc[8] = {};
    float mrow[8], lrow[8];
#pragma unroll
    for (int r = 0; r < 8; ++r) { mrow[r] = -1e30f; lrow[r] = 0.0f; }

    const float SCALE = 0.08838834764831845f;  // 1/sqrt(128)

    for (int j0 = 0; j0 <= t0 + 15; j0 += 32) {
        // ---- S tile: 16 x 32 scores ----
        v8f s0 = {}, s1 = {};
#pragma unroll
        for (int kk = 0; kk < 128; kk += 32) {
            v16h aq = load_a_frag(qbase, 2048, t0 + lo, kk, hi);
            v16h b0, b1;
#pragma unroll
            for (int s = 0; s < 16; ++s) {
                const int krow = kk + hi * 8 + (s < 8 ? s : s + 8);
                b0[s] = ktb[(size_t)krow * 1024 + j0 + lo];
                b1[s] = ktb[(size_t)krow * 1024 + j0 + 16 + lo];
            }
            s0 = WMMA_F16(aq, b0, s0);
            s1 = WMMA_F16(aq, b1, s1);
        }
        // ---- online softmax (rows live on 16-lane groups; vgpr r = row r+8*hi) ----
#pragma unroll
        for (int r = 0; r < 8; ++r) {
            const int i = t0 + r + hi * 8;
            float e0 = (j0 + lo      <= i) ? s0[r] * SCALE : -1e30f;
            float e1 = (j0 + 16 + lo <= i) ? s1[r] * SCALE : -1e30f;
            float mx = fmaxf(e0, e1);
#pragma unroll
            for (int off = 1; off < 16; off <<= 1)
                mx = fmaxf(mx, __shfl_xor(mx, off, 16));
            const float mnew = fmaxf(mrow[r], mx);
            const float p0 = __expf(e0 - mnew);
            const float p1 = __expf(e1 - mnew);
            float sum = p0 + p1;
#pragma unroll
            for (int off = 1; off < 16; off <<= 1)
                sum += __shfl_xor(sum, off, 16);
            const float corr = __expf(mrow[r] - mnew);
            lrow[r] = lrow[r] * corr + sum;
            mrow[r] = mnew;
#pragma unroll
            for (int dt = 0; dt < 8; ++dt) acc[dt][r] *= corr;
            pl[(r + hi * 8) * 40 + lo]      = (_Float16)p0;
            pl[(r + hi * 8) * 40 + 16 + lo] = (_Float16)p1;
        }
        __syncthreads();  // LDS RAW within the wave (DScnt ordering)
        // ---- reload P as a 16x32 A-fragment ----
        union { v16h v; v8h h2[2]; } pa;
        pa.h2[0] = *(const v8h*)(pl + lo * 40 + hi * 8);
        pa.h2[1] = *(const v8h*)(pl + lo * 40 + 16 + hi * 8);
        __syncthreads();  // protect LDS WAR vs next iteration
        // ---- acc += P V  (8 d-tiles) ----
#pragma unroll
        for (int dt = 0; dt < 8; ++dt) {
            v16h bvf;
#pragma unroll
            for (int s = 0; s < 16; ++s)
                bvf[s] = vb[(size_t)(j0 + hi * 8 + (s < 8 ? s : s + 8)) * 512 + dt * 16 + lo];
            acc[dt] = WMMA_F16(pa.v, bvf, acc[dt]);
        }
    }

    // ---- normalize + store y (natural (b,t,h,d) layout = (b,t,2048)) ----
#pragma unroll
    for (int dt = 0; dt < 8; ++dt) {
#pragma unroll
        for (int r = 0; r < 8; ++r) {
            const int i = t0 + r + hi * 8;
            const float val = acc[dt][r] / lrow[r];
            yh[(size_t)(b * 1024 + i) * 2048 + h * 128 + dt * 16 + lo] = (_Float16)val;
        }
    }
}

// ---------------------------------------------------------------------------
// Host launcher
// ---------------------------------------------------------------------------
extern "C" void kernel_launch(void* const* d_in, const int* in_sizes, int n_in,
                              void* d_out, int out_size, void* d_ws, size_t ws_size,
                              hipStream_t stream) {
    (void)in_sizes; (void)n_in; (void)out_size; (void)ws_size;
    const float* x  = (const float*)d_in[0];
    const float* Wq = (const float*)d_in[1];
    const float* bq = (const float*)d_in[2];
    const float* Wk = (const float*)d_in[3];
    const float* bk = (const float*)d_in[4];
    const float* Wv = (const float*)d_in[5];
    const float* bv = (const float*)d_in[6];
    const float* Wo = (const float*)d_in[7];
    const float* bo = (const float*)d_in[8];

    // Workspace layout (f16, all MB-aligned):
    char* ws = (char*)d_ws;
    _Float16* xh  = (_Float16*)(ws);                        // 4096x2048  (16 MB)
    _Float16* Wqt = (_Float16*)(ws + (16ull << 20));        // 2048x2048  ( 8 MB)
    _Float16* Wkt = (_Float16*)(ws + (24ull << 20));        // 2048x512   ( 2 MB)
    _Float16* Wvt = (_Float16*)(ws + (26ull << 20));        // 2048x512   ( 2 MB)
    _Float16* Wot = (_Float16*)(ws + (28ull << 20));        // 2048x2048  ( 8 MB)
    _Float16* qh  = (_Float16*)(ws + (36ull << 20));        // 4096x2048  (16 MB)
    _Float16* ktb = (_Float16*)(ws + (52ull << 20));        // 16x128x1024 ( 4 MB)
    _Float16* vhb = (_Float16*)(ws + (56ull << 20));        // 4096x512   ( 4 MB)
    _Float16* yh  = (_Float16*)(ws + (60ull << 20));        // 4096x2048  (16 MB)
    // total: 76 MB

    const int NX4 = 4096 * 2048 / 4;
    cvt_f32_to_f16_v4<<<(NX4 + 255) / 256, 256, 0, stream>>>(x, xh, NX4);

    dim3 tb(32, 8);
    transpose_cvt_f16_tiled<<<dim3(2048 / 32, 2048 / 32), tb, 0, stream>>>(Wq, Wqt, 2048, 2048);
    transpose_cvt_f16_tiled<<<dim3(2048 / 32,  512 / 32), tb, 0, stream>>>(Wk, Wkt, 512, 2048);
    transpose_cvt_f16_tiled<<<dim3(2048 / 32,  512 / 32), tb, 0, stream>>>(Wv, Wvt, 512, 2048);
    transpose_cvt_f16_tiled<<<dim3(2048 / 32, 2048 / 32), tb, 0, stream>>>(Wo, Wot, 2048, 2048);

    // Q = rope(x Wq^T + bq) -> qh (f16, row-major)
    gemm_wmma<1><<<dim3(2048 / 64, 4096 / 64), 32, 0, stream>>>(xh, Wqt, bq, qh, 4096, 2048, 2048);
    // K = rope(x Wk^T + bk) -> ktb transposed [b,kvh,d,t]
    gemm_wmma<2><<<dim3(512 / 64, 4096 / 64), 32, 0, stream>>>(xh, Wkt, bk, ktb, 4096, 512, 2048);
    // V = x Wv^T + bv -> vhb (f16, row-major)
    gemm_wmma<0><<<dim3(512 / 64, 4096 / 64), 32, 0, stream>>>(xh, Wvt, bv, vhb, 4096, 512, 2048);

    // causal GQA flash attention -> yh
    attn_wmma<<<dim3(64, 16, 4), 32, 0, stream>>>(qh, ktb, vhb, yh);

    // out = y Wo^T + bo (f32)
    gemm_wmma<3><<<dim3(2048 / 64, 4096 / 64), 32, 0, stream>>>(yh, Wot, bo, d_out, 4096, 2048, 2048);
}
